// FineGrainedMoE_89584427860286
// MI455X (gfx1250) — compile-verified, matrix operand order checked by reading, not run on hardware
//
#include <hip/hip_runtime.h>
#include <hip/hip_bf16.h>

// ---------------------------------------------------------------------------
// FineGrainedMoE for MI455X (gfx1250, wave32, WMMA bf16)
// B=2048 tokens, D=1024, E=32 experts, H=1024, topk=8
// ---------------------------------------------------------------------------
#define TOKENS 2048
#define DMODEL 1024
#define NEXP   32
#define FFNH   1024
#define TOPK   8

typedef __attribute__((ext_vector_type(16))) __bf16        v16bf;
typedef __attribute__((ext_vector_type(8)))  float         v8f;
typedef __attribute__((ext_vector_type(4)))  unsigned int  v4u;

union Frag16 { v16bf v; v4u q[2]; unsigned int u[8]; };

__device__ __forceinline__ unsigned short f2bf(float f) {
  unsigned int u = __float_as_uint(f);
  u += 0x7FFFu + ((u >> 16) & 1u);          // round-to-nearest-even
  return (unsigned short)(u >> 16);
}
__device__ __forceinline__ unsigned int packbf2(float a, float b) {
  return (unsigned int)f2bf(a) | ((unsigned int)f2bf(b) << 16);
}

// Fragment-major swizzle shared by A and B operands of v_wmma_f32_16x16x32_bf16:
// within one 16x32 (or 32x16) k-tile stored as u32[256] (1 KB):
//   slot(kl, row, ...) : lane = row + 16*((kl>>3)&1),
//                        dword j = ((kl&7)>>1) + ((kl>>4)&1)*4,  halfword = kl&1
__device__ __forceinline__ int frag_dword(int klocal, int row16) {
  int j    = ((klocal & 7) >> 1) + ((klocal >> 4) & 1) * 4;
  int lane = row16 + (((klocal >> 3) & 1) << 4);
  return lane * 8 + j;
}

// ---------------------------------------------------------------------------
// Kernel 1: gating. One wave32 per token, lane == expert (E==32).
// ---------------------------------------------------------------------------
__global__ void moe_gate_kernel(const float* __restrict__ x,
                                const float* __restrict__ Wg,
                                float* __restrict__ gates,
                                int* __restrict__ idxs)
{
  const int gid  = blockIdx.x * blockDim.x + threadIdx.x;
  const int tok  = gid >> 5;
  const int lane = threadIdx.x & 31;
  if (tok >= TOKENS) return;

  const float* xr = x + (size_t)tok * DMODEL;
  float acc = 0.f;
  for (int d = 0; d < DMODEL; d += 4) {
    float4 xv = *(const float4*)(xr + d);              // broadcast across wave
    acc = fmaf(xv.x, Wg[(d + 0) * NEXP + lane], acc);  // coalesced over lanes
    acc = fmaf(xv.y, Wg[(d + 1) * NEXP + lane], acc);
    acc = fmaf(xv.z, Wg[(d + 2) * NEXP + lane], acc);
    acc = fmaf(xv.w, Wg[(d + 3) * NEXP + lane], acc);
  }
  // softmax across the 32 lanes (wave32)
  float mx = acc;
  for (int off = 16; off; off >>= 1) mx = fmaxf(mx, __shfl_xor(mx, off, 32));
  float ex = __expf(acc - mx);
  float sm = ex;
  for (int off = 16; off; off >>= 1) sm += __shfl_xor(sm, off, 32);
  const float prob = ex / sm;

  // iterative top-8 (lowest index wins ties, like jax.lax.top_k)
  float p = prob;
  for (int k = 0; k < TOPK; ++k) {
    float v = p;
    for (int off = 16; off; off >>= 1) v = fmaxf(v, __shfl_xor(v, off, 32));
    unsigned long long mb = __ballot(p == v);
    int src = __ffsll(mb) - 1;
    if (lane == src) {
      gates[tok * TOPK + k] = prob;
      idxs[tok * TOPK + k]  = lane;
      p = -1.f;
    }
  }
}

// ---------------------------------------------------------------------------
// Kernel 2: deterministic per-expert compaction (one wave per expert).
// entry = token*8 + slot  (slot = position in the token's top-8 list)
// ---------------------------------------------------------------------------
__global__ void moe_compact_kernel(const int* __restrict__ idxs,
                                   int* __restrict__ counts,
                                   int* __restrict__ entries)
{
  const int e    = blockIdx.x;
  const int lane = threadIdx.x;
  int cnt = 0;
  for (int base = 0; base < TOKENS; base += 32) {
    int token = base + lane;
    int sel = -1;
#pragma unroll
    for (int k = 0; k < TOPK; ++k)
      if (idxs[token * TOPK + k] == e) sel = k;
    unsigned int mask = (unsigned int)__ballot(sel >= 0);
    int my = __popc(mask & ((1u << lane) - 1u));
    if (sel >= 0) entries[e * TOKENS + cnt + my] = token * TOPK + sel;
    cnt += __popc(mask);
  }
  if (lane == 0) counts[e] = cnt;
}

// ---------------------------------------------------------------------------
// Kernel 3 helper: 16xN GEMM (N=1024, K=1024) with bf16 WMMA.
// A: LDS fragment-major bf16, 32 k-tiles x 1 KB (lane-contiguous -> 2x b128).
// W: global f32 [1024][1024], staged per K=32 slab into Ck pre-swizzled into
//    the B-operand VGPR layout (also lane-contiguous -> 2x b128 per frag).
// Each of 8 waves accumulates 8 16x16 N-tiles (v8f each).
// ---------------------------------------------------------------------------
__device__ __forceinline__ void run_gemm32(
    const unsigned int* __restrict__ AsrcU32,   // 8192 u32 (32 tiles * 256)
    const float* __restrict__ Wglob,
    unsigned int* __restrict__ Ck,              // 16384 u32 (64 tiles * 256)
    int t, int lane, int wave,
    v8f* acc)
{
#pragma unroll
  for (int i = 0; i < 8; ++i)
#pragma unroll
    for (int r = 0; r < 8; ++r) acc[i][r] = 0.f;

  for (int kc = 0; kc < DMODEL / 32; ++kc) {
    const int kbase = kc * 32;
    // --- stage one 32x1024 slab of W into swizzled bf16 fragments ---
#pragma unroll 4
    for (int iter = 0; iter < 64; ++iter) {
      int id = t + iter * 256;          // 16 k-pairs x 1024 n
      int n  = id & 1023;
      int k  = (id >> 10) * 2;
      const float* wp = Wglob + (size_t)(kbase + k) * 1024 + n;
      float f0 = wp[0];
      float f1 = wp[1024];
      if (kc + 1 < DMODEL / 32)
        __builtin_prefetch(wp + 32 * 1024, 0, 1);   // global_prefetch_b8: next slab
      Ck[(n >> 4) * 256 + frag_dword(k, n & 15)] = packbf2(f0, f1);
    }
    __syncthreads();

    // --- A fragment (16x32 bf16): contiguous 32B per lane -> 2x ds_load_b128 ---
    Frag16 a;
    {
      const v4u* ap = (const v4u*)(AsrcU32 + kc * 256 + lane * 8);
      a.q[0] = ap[0];
      a.q[1] = ap[1];
    }
    // --- 8 N-tiles per wave ---
#pragma unroll
    for (int i = 0; i < 8; ++i) {
      Frag16 b;
      const v4u* bp = (const v4u*)(Ck + (wave * 8 + i) * 256 + lane * 8);
      b.q[0] = bp[0];
      b.q[1] = bp[1];
      acc[i] = __builtin_amdgcn_wmma_f32_16x16x32_bf16(
          false, a.v, false, b.v, (short)0, acc[i], false, false);
    }
    __syncthreads();
  }
}

// ---------------------------------------------------------------------------
// Kernel 3: sparse expert FFN. Block = (expert e, 16-token tile).
//   h = relu(Xg @ W1[e] + b1[e]);  contrib[token*8+slot] = gate*(h@W2[e]+b2[e])
// ---------------------------------------------------------------------------
__global__ __launch_bounds__(256)
void moe_ffn_kernel(const float* __restrict__ x,
                    const float* __restrict__ W1, const float* __restrict__ b1,
                    const float* __restrict__ W2, const float* __restrict__ b2,
                    const float* __restrict__ gates,
                    const int* __restrict__ entries,
                    const int* __restrict__ counts,
                    float* __restrict__ contrib)
{
  extern __shared__ unsigned char smem[];
  unsigned int* Xg = (unsigned int*)smem;     // 8192 u32: 32 frag-tiles (32 KB)
  unsigned int* Hs = Xg + 8192;               // 8192 u32: 32 frag-tiles (32 KB)
  unsigned int* Ck = Hs + 8192;               // 16384 u32 chunk (64 KB)
  float* g16  = (float*)(Ck + 16384);         // 16 gate values
  int*   cr16 = (int*)(g16 + 16);             // 16 contrib rows
  int*   tk16 = cr16 + 16;                    // 16 token ids

  const int e    = blockIdx.y;
  const int tile = blockIdx.x;
  const int cnt  = counts[e];
  if (tile * 16 >= cnt) return;

  const int t    = threadIdx.x;
  const int lane = t & 31;
  const int wave = t >> 5;
  const int nl   = lane & 15;
  const int hi   = lane >> 4;

  if (t < 16) {
    int row = tile * 16 + t;
    if (row < cnt) {
      int p   = entries[e * TOKENS + row];
      g16[t]  = gates[p];
      cr16[t] = p;              // contrib row == token*8 + slot == p
      tk16[t] = p >> 3;
    } else { g16[t] = 0.f; cr16[t] = -1; tk16[t] = 0; }
  }
  __syncthreads();

  // gather + convert 16 token rows of x into A-fragment-major bf16 LDS
  for (int iter = 0; iter < 32; ++iter) {
    int id  = t + iter * 256;        // 8192 dword writes
    int row = id >> 9;               // m = 0..15
    int k   = (id & 511) * 2;        // even k along D
    const float* xr = x + (size_t)tk16[row] * DMODEL + k;
    Xg[(k >> 5) * 256 + frag_dword(k & 31, row)] = packbf2(xr[0], xr[1]);
  }
  __syncthreads();

  v8f acc[8];

  // ---- phase A: h = relu(Xg @ W1[e] + b1[e]) -> Hs (bf16, frag-major) ----
  run_gemm32(Xg, W1 + (size_t)e * DMODEL * FFNH, Ck, t, lane, wave, acc);
  {
    const float* b1e = b1 + (size_t)e * FFNH;
    unsigned short* HsU16 = (unsigned short*)Hs;
#pragma unroll
    for (int i = 0; i < 8; ++i) {
      int n = (wave * 8 + i) * 16 + nl;       // C/D layout: lane -> n, vgpr r -> m
      float bv = b1e[n];
      int base = (n >> 5) * 256;              // k-tile along H
      int kl   = n & 31;
#pragma unroll
      for (int r = 0; r < 8; ++r) {
        int m = r + 8 * hi;
        float v = acc[i][r] + bv;
        HsU16[(base + frag_dword(kl & ~1, m)) * 2 + (kl & 1)] =
            f2bf(v > 0.f ? v : 0.f);
      }
    }
  }
  __syncthreads();

  // ---- phase B: contrib = gate * (Hs @ W2[e] + b2[e]) ----
  run_gemm32(Hs, W2 + (size_t)e * FFNH * DMODEL, Ck, t, lane, wave, acc);
  {
    const float* b2e = b2 + (size_t)e * DMODEL;
#pragma unroll
    for (int i = 0; i < 8; ++i) {
      int n = (wave * 8 + i) * 16 + nl;
      float bv = b2e[n];
#pragma unroll
      for (int r = 0; r < 8; ++r) {
        int m = r + 8 * hi;
        int crow = cr16[m];
        if (crow >= 0)
          contrib[(size_t)crow * DMODEL + n] = (acc[i][r] + bv) * g16[m];
      }
    }
  }
}

// ---------------------------------------------------------------------------
// Kernel 4: out = x + sum_k contrib[b*8+k]   (fixed order -> deterministic)
// ---------------------------------------------------------------------------
__global__ void moe_combine_kernel(const float* __restrict__ x,
                                   const float* __restrict__ contrib,
                                   float* __restrict__ out)
{
  const int b = blockIdx.x;
  const int c = threadIdx.x * 4;
  float4 s = *(const float4*)(x + (size_t)b * DMODEL + c);
#pragma unroll
  for (int k = 0; k < TOPK; ++k) {
    float4 cv = *(const float4*)(contrib + ((size_t)b * TOPK + k) * DMODEL + c);
    s.x += cv.x; s.y += cv.y; s.z += cv.z; s.w += cv.w;
  }
  *(float4*)(out + (size_t)b * DMODEL + c) = s;
}

// ---------------------------------------------------------------------------
extern "C" void kernel_launch(void* const* d_in, const int* in_sizes, int n_in,
                              void* d_out, int out_size, void* d_ws, size_t ws_size,
                              hipStream_t stream) {
  (void)in_sizes; (void)n_in; (void)out_size; (void)ws_size;
  const float* x  = (const float*)d_in[0];
  const float* Wg = (const float*)d_in[1];
  const float* W1 = (const float*)d_in[2];
  const float* b1 = (const float*)d_in[3];
  const float* W2 = (const float*)d_in[4];
  const float* b2 = (const float*)d_in[5];
  // d_in[6] = topk (hardcoded to 8 to match reference)
  float* out = (float*)d_out;

  // workspace carve-up (256B aligned)
  char* ws = (char*)d_ws;
  size_t off = 0;
  auto carve = [&](size_t bytes) {
    void* p = ws + off;
    off = (off + bytes + 255) & ~(size_t)255;
    return p;
  };
  float* gates   = (float*)carve((size_t)TOKENS * TOPK * 4);          // 64 KB
  int*   idxs    = (int*)  carve((size_t)TOKENS * TOPK * 4);          // 64 KB
  int*   counts  = (int*)  carve((size_t)NEXP * 4);
  int*   entries = (int*)  carve((size_t)NEXP * TOKENS * 4);          // 256 KB
  float* contrib = (float*)carve((size_t)TOKENS * TOPK * DMODEL * 4); // 64 MB

  moe_gate_kernel<<<dim3(TOKENS * 32 / 256), dim3(256), 0, stream>>>(x, Wg, gates, idxs);
  moe_compact_kernel<<<dim3(NEXP), dim3(32), 0, stream>>>(idxs, counts, entries);

  const size_t lds_bytes = (8192 + 8192 + 16384) * 4 + 16 * 4 * 3;    // 131264 B
  moe_ffn_kernel<<<dim3(TOKENS / 16, NEXP), dim3(256), lds_bytes, stream>>>(
      x, W1, b1, W2, b2, gates, entries, counts, contrib);

  moe_combine_kernel<<<dim3(TOKENS), dim3(256), 0, stream>>>(x, contrib, out);
}
